// TopKGate_81990925680844
// MI455X (gfx1250) — compile-verified, matrix operand order checked by reading, not run on hardware
//
#include <hip/hip_runtime.h>

// ---------------- problem constants ----------------
constexpr int SS = 65536;   // tokens
constexpr int MM = 256;     // model dim
constexpr int EE = 8;       // experts
constexpr int HH = 1024;    // hidden
constexpr int CAPC = 20480; // capacity per expert (2 * 1.25 * S/E)

// FFN tiling
constexpr int RP = 264;     // A-tile row pitch in ushorts (256 + 8 pad)
constexpr int HP = 40;      // h-scratch row pitch in ushorts (32 + 8 pad)

typedef __attribute__((ext_vector_type(16))) __bf16 v16bf;
typedef __attribute__((ext_vector_type(8)))  float  v8f;

union Frag {
    v16bf bf;
    uint4 q[2];
};

__device__ __forceinline__ v8f wmma_bf16(const Frag& a, const Frag& b, v8f c) {
    return __builtin_amdgcn_wmma_f32_16x16x32_bf16(
        false, a.bf, false, b.bf, (short)0, c, false, false);
}

__device__ __forceinline__ unsigned short f2bf(float f) {
    unsigned u = __float_as_uint(f);
    u += 0x7FFFu + ((u >> 16) & 1u);     // round to nearest even
    return (unsigned short)(u >> 16);
}
__device__ __forceinline__ float bf2f(unsigned short h) {
    return __uint_as_float(((unsigned)h) << 16);
}
__device__ __forceinline__ int kof(int i, int lane) {
    // 16-bit A-matrix 16x32 K position for packed element i of a lane
    return i + ((i >= 8) ? 8 : 0) + ((lane >= 16) ? 8 : 0);
}

// ---------------- conversion kernels ----------------
// x -> bf16 hi/lo, row-major (S x M)
__global__ __launch_bounds__(256) void conv_x(const float* __restrict__ x,
                                              unsigned short* __restrict__ xh,
                                              unsigned short* __restrict__ xl) {
    int t = blockIdx.x * 256 + threadIdx.x;
    int base = t * 4;
    #pragma unroll
    for (int j = 0; j < 4; ++j) {
        float f = x[base + j];
        unsigned short h = f2bf(f);
        unsigned short l = f2bf(f - bf2f(h));
        xh[base + j] = h;
        xl[base + j] = l;
    }
}

// w1 (E,M,H) -> pre-swizzled B fragments [e][nt(64)][kt(8)][lane(32)][i(16)]
__global__ __launch_bounds__(256) void conv_w1(const float* __restrict__ w1,
                                               unsigned short* __restrict__ wh,
                                               unsigned short* __restrict__ wl) {
    int idx = blockIdx.x * 256 + threadIdx.x;   // 2,097,152 total
    int i    = idx & 15;
    int lane = (idx >> 4) & 31;
    int kt   = (idx >> 9) & 7;
    int nt   = (idx >> 12) & 63;
    int e    = idx >> 18;
    int k = kt * 32 + kof(i, lane);
    int n = nt * 16 + (lane & 15);
    float f = w1[((size_t)e * MM + k) * HH + n];
    unsigned short h = f2bf(f);
    wh[idx] = h;
    wl[idx] = f2bf(f - bf2f(h));
}

// w2 (E,H,M) -> pre-swizzled B fragments [e][nt(16)][kt(32)][lane(32)][i(16)]
__global__ __launch_bounds__(256) void conv_w2(const float* __restrict__ w2,
                                               unsigned short* __restrict__ wh,
                                               unsigned short* __restrict__ wl) {
    int idx = blockIdx.x * 256 + threadIdx.x;   // 2,097,152 total
    int i    = idx & 15;
    int lane = (idx >> 4) & 31;
    int kt   = (idx >> 9) & 31;
    int nt   = (idx >> 14) & 15;
    int e    = idx >> 18;
    int k = kt * 32 + kof(i, lane);
    int n = nt * 16 + (lane & 15);
    float f = w2[((size_t)e * HH + k) * MM + n];
    unsigned short h = f2bf(f);
    wh[idx] = h;
    wl[idx] = f2bf(f - bf2f(h));
}

// ---------------- init kernels ----------------
__global__ __launch_bounds__(256) void init_slots(int* __restrict__ slot_token,
                                                  float* __restrict__ slot_wt) {
    int idx = blockIdx.x * 256 + threadIdx.x;
    if (idx < EE * CAPC) { slot_token[idx] = -1; slot_wt[idx] = 0.f; }
}

__global__ __launch_bounds__(256) void init_y(float* __restrict__ y) {
    int t = blockIdx.x * 256 + threadIdx.x;
    ((float4*)y)[t] = make_float4(0.f, 0.f, 0.f, 0.f);
}

// ---------------- gate: logits, softmax, top2, per-block stats ----------------
__global__ __launch_bounds__(256) void gate_kernel(const float* __restrict__ x,
                                                   const float* __restrict__ wg,
                                                   int* __restrict__ topi,
                                                   float* __restrict__ topv,
                                                   int* __restrict__ blk_cnt,
                                                   float* __restrict__ blk_gsum) {
    __shared__ float wgs[EE * MM];        // 8 KB
    __shared__ float xt[256 * 33];        // 33.8 KB (padded tile)
    __shared__ float gbuf[EE * 256];      // 8 KB
    __shared__ int   cnt[16];
    int tid = threadIdx.x;
    for (int i = tid; i < EE * MM; i += 256) wgs[i] = wg[i];
    if (tid < 16) cnt[tid] = 0;
    __syncthreads();

    int s = blockIdx.x * 256 + tid;
    float acc[EE];
    #pragma unroll
    for (int e = 0; e < EE; ++e) acc[e] = 0.f;

    for (int c = 0; c < MM; c += 32) {
        for (int i = tid; i < 256 * 32; i += 256) {
            int r = i >> 5, cc = i & 31;
            xt[r * 33 + cc] = x[(size_t)(blockIdx.x * 256 + r) * MM + c + cc];
        }
        __syncthreads();
        #pragma unroll 8
        for (int m = 0; m < 32; ++m) {
            float xv = xt[tid * 33 + m];
            #pragma unroll
            for (int e = 0; e < EE; ++e) acc[e] += xv * wgs[e * MM + c + m];
        }
        __syncthreads();
    }

    // softmax over 8
    float mx = acc[0];
    #pragma unroll
    for (int e = 1; e < EE; ++e) mx = fmaxf(mx, acc[e]);
    float g[EE], den = 0.f;
    #pragma unroll
    for (int e = 0; e < EE; ++e) { g[e] = expf(acc[e] - mx); den += g[e]; }
    float inv = 1.f / den;
    #pragma unroll
    for (int e = 0; e < EE; ++e) g[e] *= inv;

    // top-2 (earliest index wins ties, matching lax.top_k)
    int i0 = 0; float v0 = g[0];
    #pragma unroll
    for (int e = 1; e < EE; ++e) if (g[e] > v0) { v0 = g[e]; i0 = e; }
    int i1 = -1; float v1 = -1.f;
    #pragma unroll
    for (int e = 0; e < EE; ++e) if (e != i0 && g[e] > v1) { v1 = g[e]; i1 = e; }

    topi[s * 2] = i0; topi[s * 2 + 1] = i1;
    topv[s * 2] = v0; topv[s * 2 + 1] = v1;

    atomicAdd(&cnt[i0 * 2], 1);          // integer: deterministic
    atomicAdd(&cnt[i1 * 2 + 1], 1);
    #pragma unroll
    for (int e = 0; e < EE; ++e) gbuf[e * 256 + tid] = g[e];
    __syncthreads();

    if (tid < 8) {                        // deterministic ordered reduction
        float t = 0.f;
        for (int r = 0; r < 256; ++r) t += gbuf[tid * 256 + r];
        blk_gsum[blockIdx.x * 8 + tid] = t;
    }
    if (tid < 16) blk_cnt[blockIdx.x * 16 + tid] = cnt[tid];
}

// ---------------- scan: block offsets per (e,k), totals, l_loss ----------------
__global__ __launch_bounds__(256) void scan_kernel(const int* __restrict__ blk_cnt,
                                                   const float* __restrict__ blk_gsum,
                                                   int* __restrict__ blk_off,
                                                   float* __restrict__ out_loss) {
    __shared__ int   tot[16];
    __shared__ float gs[EE];
    int tid = threadIdx.x;
    if (tid < 16) {
        int t = 0;
        for (int b = 0; b < 256; ++b) t += blk_cnt[b * 16 + tid];
        tot[tid] = t;
    } else if (tid < 24) {
        int e = tid - 16; float t = 0.f;
        for (int b = 0; b < 256; ++b) t += blk_gsum[b * 8 + e];
        gs[e] = t;
    }
    __syncthreads();
    if (tid < 16) {
        int e = tid >> 1, k = tid & 1;
        int run = (k == 1) ? tot[e * 2] : 0;   // k=1 stream offset by total k=0 count
        for (int b = 0; b < 256; ++b) {
            blk_off[b * 16 + tid] = run;
            run += blk_cnt[b * 16 + tid];
        }
    }
    if (tid == 0) {
        float l = 0.f;
        const float invS = 1.f / (float)SS;
        for (int e = 0; e < EE; ++e)
            l += (gs[e] * invS) * ((float)tot[e * 2] * invS);
        *out_loss = l * (float)EE;
    }
}

// ---------------- place: intra-block ranks via ballots -> slot map ----------------
__global__ __launch_bounds__(256) void place_kernel(const int* __restrict__ topi,
                                                    const float* __restrict__ topv,
                                                    const int* __restrict__ blk_off,
                                                    int* __restrict__ slot_token,
                                                    float* __restrict__ slot_wt) {
    __shared__ int wcnt[8][16];
    int tid = threadIdx.x;
    int s = blockIdx.x * 256 + tid;
    int lane = tid & 31, wv = tid >> 5;
    int i0 = topi[s * 2], i1 = topi[s * 2 + 1];

    unsigned bal0[EE], bal1[EE];
    #pragma unroll
    for (int e = 0; e < EE; ++e) {
        bal0[e] = (unsigned)__ballot(i0 == e);
        bal1[e] = (unsigned)__ballot(i1 == e);
    }
    if (lane < 16) {
        int e = lane >> 1, k = lane & 1;
        wcnt[wv][lane] = __popc(k ? bal1[e] : bal0[e]);
    }
    __syncthreads();

    unsigned lt = (1u << lane) - 1u;
    int rank0 = __popc(bal0[i0] & lt);
    int rank1 = __popc(bal1[i1] & lt);
    int woff0 = 0, woff1 = 0;
    for (int w = 0; w < wv; ++w) { woff0 += wcnt[w][i0 * 2]; woff1 += wcnt[w][i1 * 2 + 1]; }

    int loc0 = blk_off[blockIdx.x * 16 + i0 * 2]     + woff0 + rank0;
    int loc1 = blk_off[blockIdx.x * 16 + i1 * 2 + 1] + woff1 + rank1;
    if (loc0 < CAPC) { int sl = i0 * CAPC + loc0; slot_token[sl] = s; slot_wt[sl] = topv[s * 2]; }
    if (loc1 < CAPC) { int sl = i1 * CAPC + loc1; slot_token[sl] = s; slot_wt[sl] = topv[s * 2 + 1]; }
}

// ---------------- fused expert FFN + combine ----------------
// block = 256 threads (8 waves), 64 capacity rows per block.
// split-bf16 GEMMs on v_wmma_f32_16x16x32_bf16; epilogue: weighted f32 atomics into y.
__global__ __launch_bounds__(256) void ffn_kernel(const unsigned short* __restrict__ xh,
                                                  const unsigned short* __restrict__ xl,
                                                  const unsigned short* __restrict__ w1h,
                                                  const unsigned short* __restrict__ w1l,
                                                  const unsigned short* __restrict__ w2h,
                                                  const unsigned short* __restrict__ w2l,
                                                  const float* __restrict__ b1,
                                                  const float* __restrict__ b2,
                                                  const int* __restrict__ slot_token,
                                                  const float* __restrict__ slot_wt,
                                                  float* __restrict__ y) {
    extern __shared__ __align__(16) unsigned short smem[];
    unsigned short* Ah = smem;                  // 64 x RP
    unsigned short* Al = Ah + 64 * RP;          // 64 x RP
    unsigned short* Hh = Al + 64 * RP;          // 4 x 16 x HP
    unsigned short* Hl = Hh + 4 * 16 * HP;      // 4 x 16 x HP
    __shared__ int   stok[64];
    __shared__ float swt[64];
    __shared__ int   anyflag;

    int tid = threadIdx.x;
    int e  = blockIdx.x / (CAPC / 64);
    int r0 = (blockIdx.x % (CAPC / 64)) * 64;

    if (tid == 0) anyflag = 0;
    if (tid < 64) {
        int t = slot_token[e * CAPC + r0 + tid];
        stok[tid] = t;
        swt[tid]  = slot_wt[e * CAPC + r0 + tid];
    }
    __syncthreads();
    if (tid < 64 && stok[tid] >= 0) atomicOr(&anyflag, 1);
    __syncthreads();
    if (!anyflag) return;                        // uniform early exit: empty tile

    // gather A tile (64 x 256) hi/lo into LDS, zero rows for empty slots
    for (int idx = tid; idx < 64 * 32; idx += 256) {
        int r = idx >> 5, c = (idx & 31) * 8;    // 8 ushorts = 16 B
        int t = stok[r];
        uint4 vh = make_uint4(0u, 0u, 0u, 0u), vl = vh;
        if (t >= 0) {
            vh = *(const uint4*)(xh + (size_t)t * MM + c);
            vl = *(const uint4*)(xl + (size_t)t * MM + c);
        }
        *(uint4*)(&Ah[r * RP + c]) = vh;
        *(uint4*)(&Al[r * RP + c]) = vl;
    }
    __syncthreads();

    int wv = tid >> 5, lane = tid & 31;
    int rg = wv >> 1;            // row group 0..3 (16 rows each)
    int half = wv & 1;           // column half within wave pair
    int lrow = lane & 15;
    int lhi  = lane >> 4;        // +8 K offset for upper lanes

    const unsigned short* Abh = &Ah[(rg * 16 + lrow) * RP];
    const unsigned short* Abl = &Al[(rg * 16 + lrow) * RP];
    unsigned short* Hbh = &Hh[rg * 16 * HP];
    unsigned short* Hbl = &Hl[rg * 16 * HP];

    v8f acc2[8];
    #pragma unroll
    for (int n = 0; n < 8; ++n)
        #pragma unroll
        for (int j = 0; j < 8; ++j) acc2[n][j] = 0.f;

    for (int hc = 0; hc < HH / 32; ++hc) {
        // ---- GEMM1: h[16 x 16] for n-tile nt1, K = 256 ----
        int nt1 = hc * 2 + half;
        v8f acc1;
        #pragma unroll
        for (int j = 0; j < 8; ++j) acc1[j] = 0.f;

        #pragma unroll
        for (int kt = 0; kt < 8; ++kt) {
            int koff = kt * 32 + lhi * 8;
            Frag a_h, a_l, b_h, b_l;
            a_h.q[0] = *(const uint4*)(Abh + koff);
            a_h.q[1] = *(const uint4*)(Abh + koff + 16);
            a_l.q[0] = *(const uint4*)(Abl + koff);
            a_l.q[1] = *(const uint4*)(Abl + koff + 16);
            const unsigned short* bp = w1h + ((size_t)(((e * 64 + nt1) * 8 + kt) * 32 + lane)) * 16;
            const unsigned short* bq = w1l + ((size_t)(((e * 64 + nt1) * 8 + kt) * 32 + lane)) * 16;
            b_h.q[0] = *(const uint4*)(bp);     b_h.q[1] = *(const uint4*)(bp + 8);
            b_l.q[0] = *(const uint4*)(bq);     b_l.q[1] = *(const uint4*)(bq + 8);
            acc1 = wmma_bf16(a_h, b_h, acc1);
            acc1 = wmma_bf16(a_h, b_l, acc1);
            acc1 = wmma_bf16(a_l, b_h, acc1);
        }

        // bias + relu -> split bf16 into h scratch
        float bias1 = b1[e * HH + nt1 * 16 + lrow];
        #pragma unroll
        for (int j = 0; j < 8; ++j) {
            float v = acc1[j] + bias1;
            v = v > 0.f ? v : 0.f;
            unsigned short h16 = f2bf(v);
            unsigned short l16 = f2bf(v - bf2f(h16));
            int row = j + 8 * lhi;
            int col = half * 16 + lrow;
            Hbh[row * HP + col] = h16;
            Hbl[row * HP + col] = l16;
        }
        __syncthreads();

        // ---- GEMM2: eo[16 x 256] += h[16 x 32] * w2[32 x 256] ----
        Frag a2h, a2l;
        int k2 = lhi * 8;
        const unsigned short* hr_h = Hbh + lrow * HP;
        const unsigned short* hr_l = Hbl + lrow * HP;
        a2h.q[0] = *(const uint4*)(hr_h + k2);      a2h.q[1] = *(const uint4*)(hr_h + k2 + 16);
        a2l.q[0] = *(const uint4*)(hr_l + k2);      a2l.q[1] = *(const uint4*)(hr_l + k2 + 16);

        #pragma unroll
        for (int n = 0; n < 8; ++n) {
            int nt2 = half * 8 + n;
            const unsigned short* bp = w2h + ((size_t)(((e * 16 + nt2) * 32 + hc) * 32 + lane)) * 16;
            const unsigned short* bq = w2l + ((size_t)(((e * 16 + nt2) * 32 + hc) * 32 + lane)) * 16;
            Frag b_h, b_l;
            b_h.q[0] = *(const uint4*)(bp);     b_h.q[1] = *(const uint4*)(bp + 8);
            b_l.q[0] = *(const uint4*)(bq);     b_l.q[1] = *(const uint4*)(bq + 8);
            acc2[n] = wmma_bf16(a2h, b_h, acc2[n]);
            acc2[n] = wmma_bf16(a2h, b_l, acc2[n]);
            acc2[n] = wmma_bf16(a2l, b_h, acc2[n]);
        }
        __syncthreads();   // protect h scratch before next chunk
    }

    // epilogue: y[token] += gate_weight * (eo + b2)   (<=2 addends/elem -> deterministic)
    #pragma unroll
    for (int n = 0; n < 8; ++n) {
        int nt2 = half * 8 + n;
        int col = nt2 * 16 + lrow;
        float bias2 = b2[e * MM + col];
        #pragma unroll
        for (int j = 0; j < 8; ++j) {
            int lr = rg * 16 + j + 8 * lhi;
            int t = stok[lr];
            if (t >= 0) {
                float v = (acc2[n][j] + bias2) * swt[lr];
                atomicAdd(&y[(size_t)t * MM + col], v);
            }
        }
    }
}

// ---------------- host ----------------
extern "C" void kernel_launch(void* const* d_in, const int* in_sizes, int n_in,
                              void* d_out, int out_size, void* d_ws, size_t ws_size,
                              hipStream_t stream) {
    const float* x  = (const float*)d_in[0];
    const float* wg = (const float*)d_in[1];
    const float* w1 = (const float*)d_in[2];
    const float* b1 = (const float*)d_in[3];
    const float* w2 = (const float*)d_in[4];
    const float* b2 = (const float*)d_in[5];
    float* y = (float*)d_out;               // [0, S*M) = y, [S*M] = l_loss

    // workspace carve (~82 MB)
    unsigned short* xh  = (unsigned short*)d_ws;
    unsigned short* xl  = xh + (size_t)SS * MM;
    unsigned short* w1h = xl + (size_t)SS * MM;
    unsigned short* w1l = w1h + 2097152;
    unsigned short* w2h = w1l + 2097152;
    unsigned short* w2l = w2h + 2097152;
    int*   slot_token = (int*)(w2l + 2097152);
    float* slot_wt    = (float*)(slot_token + EE * CAPC);
    int*   topi       = (int*)(slot_wt + EE * CAPC);
    float* topv       = (float*)(topi + 2 * SS);
    int*   blk_cnt    = (int*)(topv + 2 * SS);
    float* blk_gsum   = (float*)(blk_cnt + 256 * 16);
    int*   blk_off    = (int*)(blk_gsum + 256 * 8);

    conv_x <<<16384, 256, 0, stream>>>(x, xh, xl);
    conv_w1<<< 8192, 256, 0, stream>>>(w1, w1h, w1l);
    conv_w2<<< 8192, 256, 0, stream>>>(w2, w2h, w2l);
    init_slots<<<640, 256, 0, stream>>>(slot_token, slot_wt);
    init_y<<<16384, 256, 0, stream>>>(y);

    gate_kernel<<<256, 256, 0, stream>>>(x, wg, topi, topv, blk_cnt, blk_gsum);
    scan_kernel<<<1, 256, 0, stream>>>(blk_cnt, blk_gsum, blk_off, y + (size_t)SS * MM);
    place_kernel<<<256, 256, 0, stream>>>(topi, topv, blk_off, slot_token, slot_wt);

    size_t smem = (size_t)(64 * RP * 2 + 4 * 16 * HP * 2) * sizeof(unsigned short); // 77,824 B
    ffn_kernel<<<EE * (CAPC / 64), 256, smem, stream>>>(
        xh, xl, w1h, w1l, w2h, w2l, b1, b2, slot_token, slot_wt, y);
}
